// TwoPassSSMKernel_36335423324725
// MI455X (gfx1250) — matrix operand order repaired and checked by reading, not compiled, not run on hardware
//
#include <hip/hip_runtime.h>
#include <hip/hip_bf16.h>

// ---------------------------------------------------------------------------
// Mamba-2 chunked SSM forward for gfx1250 (MI455X), bf16 WMMA + fp32 accum.
// dt is the constant 0.1 in the reference, so all decays are per-head scalars:
//   a_h = softplus(0.1 + dt_bias[h]) * (-exp(A_log[h]))
// GEMMs: double-buffered LDS tiles staged by the Tensor Data Mover
// (tensor_load_to_lds, TENSORcnt) with an async-global->LDS fallback.
// ---------------------------------------------------------------------------

typedef __bf16 bf16;
typedef __attribute__((ext_vector_type(16))) __bf16 v16bf;
typedef __attribute__((ext_vector_type(8)))  __bf16 v8bf;
typedef __attribute__((ext_vector_type(8)))  float  v8f;
typedef __attribute__((ext_vector_type(4)))  unsigned int v4u;
typedef __attribute__((ext_vector_type(8)))  int v8i;
typedef __attribute__((ext_vector_type(4)))  int v4i;
typedef unsigned int       u32;
typedef unsigned long long u64;

#define NH    112
#define HP    64
#define NS    64
#define CHK   256
#define DM    3584
#define INNER 7168
#define XZW   14336
#define NG    2
#define BW    (NG * NS)   // 128 floats per token row of B / C
#define SPITCH 264        // padded LDS pitch (bf16) for the score bands

#define APITCH 40         // LDS pitch of A tile (128 x 32 bf16): 64B row + 16B pad
#define WPITCH 136        // LDS pitch of W tile (32 x 128 bf16): 256B row + 16B pad

#if __has_builtin(__builtin_amdgcn_tensor_load_to_lds)
#define HAVE_TDM 1
#else
#define HAVE_TDM 0
#endif

static __device__ __forceinline__ float softplus_f(float v) {
  return (v > 20.f) ? v : log1pf(__expf(v));
}
static __device__ __forceinline__ float sigmoid_f(float v) {
  return 1.f / (1.f + __expf(-v));
}

// Low 32 bits of a generic pointer to __shared__ = workgroup LDS byte offset.
static __device__ __forceinline__ u32 lds_addr32(const void* p) {
  return (u32)(unsigned long long)(uintptr_t)p;
}
// gfx1250 async DMA: 16B global -> LDS, tracked by ASYNCcnt (fallback path).
static __device__ __forceinline__ void async_ld_b128(u32 lds, const void* g) {
  asm volatile("global_load_async_to_lds_b128 %0, %1, off"
               :: "v"(lds), "v"((u64)(uintptr_t)g) : "memory");
}
static __device__ __forceinline__ void wait_async0() {
  asm volatile("s_wait_asynccnt 0x0" ::: "memory");
}

#if HAVE_TDM
// One 2D TDM descriptor: tile (tile_d1 rows x tile_d0 elems) of a bf16 tensor
// whose dim0 line length is stride0 elems; LDS rows padded per pad codes.
// D# bit layout per CDNA5 ISA.
static __device__ __forceinline__ void tdm_load_2d(
    u32 lds_off, const void* gaddr, u32 tensor_d0, u32 tensor_d1,
    u32 tile_d0, u32 tile_d1, u64 stride0, u32 pad_int_code, u32 pad_amt_code) {
  const u64 ga = (u64)(uintptr_t)gaddr;
  v4u g0;
  g0[0] = 1u;                                            // count=1, user mode
  g0[1] = lds_off;                                       // lds_addr
  g0[2] = (u32)ga;                                       // global_addr[31:0]
  g0[3] = (u32)((ga >> 32) & 0x01FFFFFFu) | 0x80000000u; // [56:32] | type=2
  v8i g1;
  g1[0] = (int)((1u << 16)                // data_size = 1 (2 bytes)
              | (1u << 20)                // pad_enable
              | (pad_int_code << 22) | (pad_amt_code << 25));
  g1[1] = (int)((tensor_d0 & 0xFFFFu) << 16);            // abar=0 | td0.lo
  g1[2] = (int)((tensor_d0 >> 16) | ((tensor_d1 & 0xFFFFu) << 16));
  g1[3] = (int)((tensor_d1 >> 16) | (tile_d0 << 16));
  g1[4] = (int)tile_d1;                                  // tile_d2 = 0
  g1[5] = (int)(u32)stride0;
  g1[6] = (int)(u32)((stride0 >> 32) & 0xFFFFu);         // dim1_stride = 0
  g1[7] = 0;
  v4i z4 = {0, 0, 0, 0};
  v8i z8 = {0, 0, 0, 0, 0, 0, 0, 0};
  // 6-arg form (amdgpu-toolchain / clang-23 prototype)
  __builtin_amdgcn_tensor_load_to_lds(g0, g1, z4, z4, z8, 0);
}
#endif

// ----------------------------- converts ------------------------------------
__global__ void k_f32_to_bf16(const float* __restrict__ in, bf16* __restrict__ out, int n) {
  for (int i = blockIdx.x * blockDim.x + threadIdx.x; i < n; i += gridDim.x * blockDim.x)
    out[i] = (bf16)in[i];
}
// in: R x C f32 row-major  ->  out: C x R bf16 row-major (transpose)
__global__ void k_transpose_f32_bf16(const float* __restrict__ in, bf16* __restrict__ out,
                                     int R, int C) {
  for (int i = blockIdx.x * blockDim.x + threadIdx.x; i < R * C; i += gridDim.x * blockDim.x) {
    int r = i / C, c = i - r * C;
    out[(size_t)c * R + r] = (bf16)in[i];
  }
}

// ------------------------- block-tiled NN WMMA GEMM -------------------------
// C[M,N] (f32) = A[M,K] (bf16 rm) @ W[K,N] (bf16 rm).  M,N % 128, K % 32 == 0.
// Block 256 thr = 8 waves; block tile 128x128; wave tile 32x64 (2x4 wmma).
// Double-buffered LDS; tiles staged by TDM (wave 0 issues 2 descriptors per
// K-step, waits TENSORcnt) or per-lane async b128 DMA; one barrier per step:
//   wait(tile k) -> barrier -> issue stage(k+1) -> compute(k)
__global__ void k_gemm_nn(const bf16* __restrict__ A, const bf16* __restrict__ W,
                          float* __restrict__ C, int M, int N, int K) {
  __shared__ bf16 As[2][128][APITCH];
  __shared__ bf16 Ws[2][32][WPITCH];

  const int lane = threadIdx.x & 31;
  const int wave = threadIdx.x >> 5;
  const int lm   = lane & 15;
  const int hi8  = (lane >= 16) ? 8 : 0;
  const int ko   = (lane < 16) ? 0 : 8;

  const int m_base = blockIdx.y * 128;
  const int n_base = blockIdx.x * 128;
  const int m_wave = (wave & 3) * 32;   // 4 waves along M
  const int n_wave = (wave >> 2) * 64;  // 2 waves along N

  const int t  = threadIdx.x;
  const int ar = t >> 1, ac = (t & 1) << 4;         // A stage: 128 rows x 32
  const int wr = t >> 3, wc = (t & 7) << 4;         // W stage: 32 rows x 128

  v8f acc[2][4];
#pragma unroll
  for (int mt = 0; mt < 2; ++mt)
#pragma unroll
    for (int nt = 0; nt < 4; ++nt) acc[mt][nt] = (v8f){};

  const int nk = K >> 5;

  auto stage = [&](int buf, int kb) {
#if HAVE_TDM
    if (wave == 0) {
      // A tile: 128 rows x 32 elems, rows of K-long dim0; pad 64B+16B (codes 3,3)
      tdm_load_2d(lds_addr32(&As[buf][0][0]), A + (size_t)m_base * K + kb,
                  (u32)K, (u32)M, 32u, 128u, (u64)K, 3u, 3u);
      // W tile: 32 rows x 128 elems, rows of N-long dim0; pad 256B+16B (codes 5,3)
      tdm_load_2d(lds_addr32(&Ws[buf][0][0]), W + (size_t)kb * N + n_base,
                  (u32)N, (u32)K, 128u, 32u, (u64)N, 5u, 3u);
    }
#else
    const bf16* ga = A + (size_t)(m_base + ar) * K + kb + ac;
    u32 la = lds_addr32(&As[buf][ar][ac]);
    async_ld_b128(la,      ga);
    async_ld_b128(la + 16, ga + 8);
    const bf16* gw = W + (size_t)(kb + wr) * N + n_base + wc;
    u32 lw = lds_addr32(&Ws[buf][wr][wc]);
    async_ld_b128(lw,      gw);
    async_ld_b128(lw + 16, gw + 8);
    __builtin_prefetch(ga + 32, 0, 3);
    __builtin_prefetch(gw + (size_t)32 * N, 0, 3);
#endif
  };

  stage(0, 0);
  for (int ks = 0; ks < nk; ++ks) {
    const int buf = ks & 1;
#if HAVE_TDM
    if (wave == 0) __builtin_amdgcn_s_wait_tensorcnt(0);
#else
    wait_async0();
#endif
    __syncthreads();                      // tile `ks` visible to all waves;
                                          // also: everyone done reading buf^1
    if (ks + 1 < nk) stage(buf ^ 1, (ks + 1) << 5);

    // ---- fragments + 8 WMMAs on buffer `buf` ----
    union { v16bf v; v8bf h[2]; } afr[2];
#pragma unroll
    for (int mt = 0; mt < 2; ++mt) {
      const bf16* ap = &As[buf][m_wave + mt * 16 + lm][ko];
      afr[mt].h[0] = *(const v8bf*)(ap);
      afr[mt].h[1] = *(const v8bf*)(ap + 16);
    }
#pragma unroll
    for (int nt = 0; nt < 4; ++nt) {
      union { v16bf v; v8bf h[2]; } bfr;
      const bf16* bp = &Ws[buf][ko][n_wave + nt * 16 + lm];
#pragma unroll
      for (int s = 0; s < 8; ++s) { bfr.v[s] = bp[s * WPITCH]; bfr.v[s + 8] = bp[(s + 16) * WPITCH]; }
#pragma unroll
      for (int mt = 0; mt < 2; ++mt)
        acc[mt][nt] = __builtin_amdgcn_wmma_f32_16x16x32_bf16(
            false, afr[mt].v, false, bfr.v, (short)0, acc[mt][nt], false, false);
    }
  }

  // ---- writeback (D layout: VGPR r -> row +r, +8 for upper half-wave) ----
#pragma unroll
  for (int mt = 0; mt < 2; ++mt) {
    const int row0 = m_base + m_wave + mt * 16 + hi8;
#pragma unroll
    for (int nt = 0; nt < 4; ++nt) {
      float* cp = C + (size_t)row0 * N + n_base + n_wave + nt * 16 + lm;
#pragma unroll
      for (int r = 0; r < 8; ++r) cp[(size_t)r * N] = acc[mt][nt][r];
    }
  }
}

// --------------------- Xdt = bf16(x * dt_sp[head]) --------------------------
__global__ void k_xdt(const float* __restrict__ xz, const float* __restrict__ dt_bias,
                      bf16* __restrict__ xdt, int total) {
  for (int i = blockIdx.x * blockDim.x + threadIdx.x; i < total; i += gridDim.x * blockDim.x) {
    int row = i / INNER, col = i - row * INNER;
    int h = col >> 6;
    float dtsp = softplus_f(0.1f + dt_bias[h]);
    xdt[i] = (bf16)(xz[(size_t)row * XZW + col] * dtsp);
  }
}

// -------- per-chunk end states: states[k,h,p,n] (64x64 per chunk/head) ------
__global__ void k_states(const float* __restrict__ xz, const float* __restrict__ Bsrc,
                         const float* __restrict__ A_log, const float* __restrict__ dt_bias,
                         float* __restrict__ states) {
  const int kh = blockIdx.x;
  const int h = kh % NH, k = kh / NH, g = h / (NH / NG);
  const float dtsp = softplus_f(0.1f + dt_bias[h]);
  const float a = -__expf(A_log[h]) * dtsp;
  const int p  = threadIdx.x >> 2;
  const int nb = (threadIdx.x & 3) << 4;
  const int row0 = k * CHK;

  float acc[16];
#pragma unroll
  for (int i = 0; i < 16; ++i) acc[i] = 0.f;
  for (int j = 0; j < CHK; ++j) {
    float w  = __expf(a * (float)(CHK - 1 - j)) * dtsp;
    float xv = xz[(size_t)(row0 + j) * XZW + h * HP + p] * w;
    const float* br = Bsrc + (size_t)(row0 + j) * BW + g * NS + nb;
#pragma unroll
    for (int n = 0; n < 16; ++n) acc[n] = fmaf(xv, br[n], acc[n]);
  }
  float* sp = states + (size_t)kh * (HP * NS) + p * NS + nb;
#pragma unroll
  for (int n = 0; n < 16; ++n) sp[n] = acc[n];
}

// ---- inter-chunk scan: prevT[k,h,n,p] = bf16(state ENTERING chunk k) -------
__global__ void k_scan(const float* __restrict__ states, const float* __restrict__ A_log,
                       const float* __restrict__ dt_bias, bf16* __restrict__ prevT,
                       int nch) {
  const int h = blockIdx.x;
  const float dtsp = softplus_f(0.1f + dt_bias[h]);
  const float a = -__expf(A_log[h]) * dtsp;
  const float dec = __expf(a * (float)CHK);
  const int e0 = threadIdx.x * 16;
  float S[16];
#pragma unroll
  for (int i = 0; i < 16; ++i) S[i] = 0.f;
  for (int k = 0; k < nch; ++k) {
    const float* st = states + (size_t)(k * NH + h) * (HP * NS);
    bf16* pv = prevT + (size_t)(k * NH + h) * (HP * NS);
#pragma unroll
    for (int i = 0; i < 16; ++i) {
      int idx = e0 + i;
      int p = idx >> 6, n = idx & 63;
      pv[n * HP + p] = (bf16)S[i];          // transposed for WMMA B-frags
      S[i] = S[i] * dec + st[idx];
    }
  }
}

// ------------------- main per-(chunk, head) SSM kernel ----------------------
// 256 thr = 8 waves; wave w owns i-row-bands {w, w+8} (16 rows each).
// Phase 1: scores = (C B^T) .* L  (WMMA on Cb x BT, lower-tri blocks only),
//          bf16 into per-wave LDS band (D-tile layout out, A-frag layout in).
// Phase 2: Y_diag = scores @ Xdt   (WMMA)
// Phase 3: Y_off  = C @ prev^T     (WMMA), scaled by exp(a*(i+1))
// Phase 4: + x*D, sigmoid(z) gate, emit bf16 Ypre for the output GEMM.
__global__ void k_chunk(const float* __restrict__ xz, const bf16* __restrict__ BT,
                        const bf16* __restrict__ Cb, const bf16* __restrict__ xdt,
                        const bf16* __restrict__ prevT,
                        const float* __restrict__ A_log, const float* __restrict__ dt_bias,
                        const float* __restrict__ Dvec, bf16* __restrict__ ypre, int L) {
  extern __shared__ bf16 smem[];   // 8 waves * 16 rows * SPITCH bf16
  const int kh = blockIdx.x;
  const int h = kh % NH, k = kh / NH, g = h / (NH / NG);
  const float dtsp = softplus_f(0.1f + dt_bias[h]);
  const float a = -__expf(A_log[h]) * dtsp;
  const float Dh = Dvec[h];
  const int lane = threadIdx.x & 31;
  const int wave = threadIdx.x >> 5;
  const int lm = lane & 15;
  const int hi8 = (lane >= 16) ? 8 : 0;
  const int ko = (lane < 16) ? 0 : 8;
  bf16* sband = smem + wave * 16 * SPITCH;
  const int row0 = k * CHK;

  for (int bb = 0; bb < 2; ++bb) {
    const int bi = wave + bb * 8;
    const int i0 = bi << 4;

    // A-fragments of C rows [i0..i0+15] over n = 0..63 (two K-steps of 32)
    union { v16bf v; v8bf hv[2]; } ac[2];
    {
      const bf16* cp = Cb + (size_t)(row0 + i0 + lm) * BW + g * NS + ko;
#pragma unroll
      for (int ks = 0; ks < 2; ++ks) {
        ac[ks].hv[0] = *(const v8bf*)(cp + ks * 32);
        ac[ks].hv[1] = *(const v8bf*)(cp + ks * 32 + 16);
      }
    }

    // -------- phase 1: lower-triangular score blocks --------
    for (int jb = 0; jb <= bi; ++jb) {
      const int j0 = jb << 4;
      v8f s = {};
#pragma unroll
      for (int ks = 0; ks < 2; ++ks) {
        const int kb = ks * 32;
        // B-frag from transposed B: lane%16 = j column, elements over n in table order
        union { v16bf v; v8bf hv[2]; } bfrg;
        const bf16* bp = BT + (size_t)(g * NS + kb + ko) * L + row0 + j0 + lm;
#pragma unroll
        for (int s = 0; s < 8; ++s) { bfrg.v[s] = bp[(size_t)s * L]; bfrg.v[s + 8] = bp[(size_t)(s + 16) * L]; }
        s = __builtin_amdgcn_wmma_f32_16x16x32_bf16(false, ac[ks].v, false, bfrg.v,
                                                    (short)0, s, false, false);
      }
      const int j = j0 + lm;
#pragma unroll
      for (int r = 0; r < 8; ++r) {
        const int i = i0 + hi8 + r;
        float v = (j <= i) ? s[r] * __expf(a * (float)(i - j)) : 0.f;
        sband[(hi8 + r) * SPITCH + j] = (bf16)v;
      }
    }
    // zero-fill the unwritten 16x16 block that falls inside the K loop
    if ((bi & 1) == 0) {
      const int j0 = (bi + 1) << 4;
      v8bf z = {};
      *(v8bf*)&sband[lm * SPITCH + j0 + ko] = z;
    }

    // -------- phase 2 + 3: Y = S @ Xdt  and  Yoff = C @ prevT --------
    v8f zero = {};
    v8f yd[4], yo[4];
#pragma unroll
    for (int q = 0; q < 4; ++q) { yd[q] = zero; yo[q] = zero; }

    const int nk2 = (bi + 2) >> 1;          // # of 32-wide K steps over j
    for (int ks = 0; ks < nk2; ++ks) {
      const int kb = ks * 32;
      union { v16bf v; v8bf hv[2]; } af;    // scores A-frag from LDS
      const bf16* sp = sband + lm * SPITCH + kb + ko;
      af.hv[0] = *(const v8bf*)(sp);
      af.hv[1] = *(const v8bf*)(sp + 16);
#pragma unroll
      for (int q = 0; q < 4; ++q) {
        v16bf bv;
        const bf16* xp = xdt + (size_t)(row0 + kb + ko) * INNER + h * HP + q * 16 + lm;
#pragma unroll
        for (int s = 0; s < 8; ++s) { bv[s] = xp[(size_t)s * INNER]; bv[s + 8] = xp[(size_t)(s + 16) * INNER]; }
        yd[q] = __builtin_amdgcn_wmma_f32_16x16x32_bf16(false, af.v, false, bv,
                                                        (short)0, yd[q], false, false);
      }
    }

    const bf16* pvb = prevT + (size_t)kh * (HP * NS);
#pragma unroll
    for (int ks = 0; ks < 2; ++ks) {
      const int kb = ks * 32;
#pragma unroll
      for (int q = 0; q < 4; ++q) {
        v16bf bv;
        const bf16* pp = pvb + (size_t)(kb + ko) * HP + q * 16 + lm;
#pragma unroll
        for (int s = 0; s < 8; ++s) { bv[s] = pp[s * HP]; bv[s + 8] = pp[(s + 16) * HP]; }
        yo[q] = __builtin_amdgcn_wmma_f32_16x16x32_bf16(false, ac[ks].v, false, bv,
                                                        (short)0, yo[q], false, false);
      }
    }

    // -------- phase 4: combine, D-term, gate, emit bf16 --------
#pragma unroll
    for (int q = 0; q < 4; ++q) {
      const int col = h * HP + q * 16 + lm;
#pragma unroll
      for (int r = 0; r < 8; ++r) {
        const int i = i0 + hi8 + r;
        const size_t grow = (size_t)(row0 + i);
        float xv = xz[grow * XZW + col];
        float zv = xz[grow * XZW + INNER + col];
        float y = yd[q][r] + yo[q][r] * __expf(a * (float)(i + 1)) + xv * Dh;
        y *= sigmoid_f(zv);
        ypre[grow * INNER + col] = (bf16)y;
      }
    }
  }
}

// ---------------------------------------------------------------------------
extern "C" void kernel_launch(void* const* d_in, const int* in_sizes, int n_in,
                              void* d_out, int out_size, void* d_ws, size_t ws_size,
                              hipStream_t stream) {
  (void)n_in; (void)out_size; (void)ws_size;
  const float* hidden  = (const float*)d_in[0];
  const float* W_in    = (const float*)d_in[1];
  const float* W_out   = (const float*)d_in[2];
  const float* A_log   = (const float*)d_in[3];
  const float* Dvec    = (const float*)d_in[4];
  const float* dt_bias = (const float*)d_in[5];
  const float* Bsrc    = (const float*)d_in[6];
  const float* Csrc    = (const float*)d_in[7];
  float* out = (float*)d_out;

  const int L   = in_sizes[0] / DM;   // 4096
  const int nch = L / CHK;            // 16

  // ---- workspace carve (256B aligned) ----
  char* p = (char*)d_ws;
  auto carve = [&](size_t bytes) { char* r = p; p += (bytes + 255) & ~(size_t)255; return r; };
  bf16*  Hb     = (bf16*) carve((size_t)L * DM * 2);
  bf16*  WiT    = (bf16*) carve((size_t)DM * XZW * 2);     // W_in^T  [DM][XZW]
  bf16*  WoT    = (bf16*) carve((size_t)INNER * DM * 2);   // W_out^T [INNER][DM]
  bf16*  BT     = (bf16*) carve((size_t)BW * L * 2);       // B^T     [128][L]
  bf16*  Cb     = (bf16*) carve((size_t)L * BW * 2);
  float* XZ     = (float*)carve((size_t)L * XZW * 4);
  bf16*  Xdt    = (bf16*) carve((size_t)L * INNER * 2);
  float* states = (float*)carve((size_t)nch * NH * HP * NS * 4);
  bf16*  prevT  = (bf16*) carve((size_t)nch * NH * HP * NS * 2);
  bf16*  Ypre   = (bf16*) carve((size_t)L * INNER * 2);

  // ---- 1) converts / transposes ----
  {
    int n;
    n = L * DM;     k_f32_to_bf16<<<(n + 255) / 256, 256, 0, stream>>>(hidden, Hb, n);
    n = L * BW;     k_f32_to_bf16<<<(n + 255) / 256, 256, 0, stream>>>(Csrc, Cb, n);
    n = XZW * DM;   k_transpose_f32_bf16<<<(n + 255) / 256, 256, 0, stream>>>(W_in, WiT, XZW, DM);
    n = DM * INNER; k_transpose_f32_bf16<<<(n + 255) / 256, 256, 0, stream>>>(W_out, WoT, DM, INNER);
    n = L * BW;     k_transpose_f32_bf16<<<(n + 255) / 256, 256, 0, stream>>>(Bsrc, BT, L, BW);
  }

  // ---- 2) XZ = H @ W_in^T  (NN with pre-transposed weights) ----
  k_gemm_nn<<<dim3(XZW / 128, L / 128), 256, 0, stream>>>(Hb, WiT, XZ, L, XZW, DM);

  // ---- 3) Xdt prep ----
  {
    int n = L * INNER;
    k_xdt<<<(n + 255) / 256, 256, 0, stream>>>(XZ, dt_bias, Xdt, n);
  }

  // ---- 4) chunk end-states ----
  k_states<<<nch * NH, 256, 0, stream>>>(XZ, Bsrc, A_log, dt_bias, states);

  // ---- 5) inter-chunk scan ----
  k_scan<<<NH, 256, 0, stream>>>(states, A_log, dt_bias, prevT, nch);

  // ---- 6) main SSM chunk kernel ----
  {
    const int smem_bytes = 8 * 16 * SPITCH * 2;   // 67584 B (< 320KB WGP LDS)
    (void)hipFuncSetAttribute(reinterpret_cast<const void*>(k_chunk),
                              hipFuncAttributeMaxDynamicSharedMemorySize, smem_bytes);
    k_chunk<<<nch * NH, 256, smem_bytes, stream>>>(XZ, BT, Cb, Xdt, prevT,
                                                   A_log, dt_bias, Dvec, Ypre, L);
  }

  // ---- 7) out = Ypre @ W_out^T ----
  k_gemm_nn<<<dim3(DM / 128, L / 128), 256, 0, stream>>>(Ypre, WoT, out, L, DM, INNER);
}
